// DimeNetPP_original_56092272885946
// MI455X (gfx1250) — matrix operand order repaired and checked by reading, not compiled, not run on hardware
//
#include <hip/hip_runtime.h>

// ---------------------------------------------------------------------------
// DimeNet++ forward for MI455X (gfx1250, wave32, WMMA).
// All heavy GEMMs -> v_wmma_f32_16x16x32_f16 (f16 in, f32 acc).
// Segment sums -> global_atomic_add_f32 (L2-resident accumulators).
// Bessel zero tables recomputed on-device (they are Python module constants).
// ---------------------------------------------------------------------------

#define E_EDGES 65536
#define T_TRIP  524288
#define A_ATOMS 8192
#define M_MOL   256
#define EMB     128
#define OUTEMB  256
#define INTEMB  64
#define BASIS   8
#define NRAD    6
#define LMAX    7
#define NLBL    12

#define ENV_A (-28.0f)   // -(p+1)(p+2)/2, p=6
#define ENV_B ( 48.0f)   //  p(p+2)
#define ENV_C (-21.0f)   // -p(p+1)/2

// Try the gfx1250 async global->LDS path; falls back cleanly if the builtins
// are not declared by this toolchain.
#ifndef DN_TRY_ASYNC
#define DN_TRY_ASYNC 1
#endif
#if DN_TRY_ASYNC && defined(__HIP_DEVICE_COMPILE__) && \
    __has_builtin(__builtin_amdgcn_global_load_async_to_lds_b128) && \
    __has_builtin(__builtin_amdgcn_s_wait_asynccnt)
#define DN_ASYNC_OK 1
#else
#define DN_ASYNC_OK 0
#endif

typedef __attribute__((ext_vector_type(16))) _Float16 v16h;
typedef __attribute__((ext_vector_type(8)))  float    v8f;
#if DN_ASYNC_OK
typedef int dn_v4i __attribute__((vector_size(16)));
typedef __attribute__((address_space(1))) dn_v4i dn_gv4i;  // global
typedef __attribute__((address_space(3))) dn_v4i dn_lv4i;  // LDS
#endif

__device__ __forceinline__ float swishf(float v) {
  return v * (1.0f / (1.0f + __expf(-v)));
}

// spherical Bessel j_l via upward recurrence (x > 0)
__device__ float sph_jn(int l, float x) {
  float j0 = sinf(x) / x;
  if (l == 0) return j0;
  float jm = j0;
  float jc = j0 / x - cosf(x) / x;
  for (int ll = 1; ll < l; ++ll) {
    float jn = (2.0f * ll + 1.0f) / x * jc - jm;
    jm = jc; jc = jn;
  }
  return jc;
}

// --------------------------- utility: zero fill ----------------------------
__global__ void dn_zero(float* __restrict__ p, long n) {
  long i = (long)blockIdx.x * blockDim.x + threadIdx.x;
  if (i < n) p[i] = 0.0f;
}

// ------------------- Bessel zeros + norms (module constants) ---------------
__global__ void dn_bessel_init(float* __restrict__ zj, float* __restrict__ bnorm) {
  int l = threadIdx.x;
  if (l >= LMAX) return;
  int found = 0;
  float x = 0.1f;
  float fprev = sph_jn(l, x);
  const float dx = 0.005f;
  while (found < NRAD && x < 32.0f) {
    x += dx;
    float f = sph_jn(l, x);
    if (fprev * f < 0.0f) {
      float a = x - dx, b = x, fa = fprev;
      for (int it = 0; it < 50; ++it) {
        float m = 0.5f * (a + b);
        float fm = sph_jn(l, m);
        if (fa * fm <= 0.0f) { b = m; } else { a = m; fa = fm; }
      }
      float z = 0.5f * (a + b);
      zj[l * NRAD + found] = z;
      bnorm[l * NRAD + found] = sqrtf(2.0f) / fabsf(sph_jn(l + 1, z));
      ++found;
    }
    fprev = f;
  }
}

// -------------------------- edge radial bases ------------------------------
__global__ __launch_bounds__(256) void dn_edge_basis(
    const float* __restrict__ R, const int* __restrict__ ii,
    const int* __restrict__ jj, const float* __restrict__ freq,
    const float* __restrict__ zj, const float* __restrict__ bnorm,
    float* __restrict__ rbf, float* __restrict__ rbfenv) {
  long e = (long)blockIdx.x * 256 + threadIdx.x;
  if (e >= E_EDGES) return;
  int i = ii[e], j = jj[e];
  float dx = R[i * 3 + 0] - R[j * 3 + 0];
  float dy = R[i * 3 + 1] - R[j * 3 + 1];
  float dz = R[i * 3 + 2] - R[j * 3 + 2];
  float D = sqrtf(fmaxf(dx * dx + dy * dy + dz * dz, 0.0f));
  float d = D * 0.2f;  // / CUTOFF
  float d2 = d * d, d4 = d2 * d2, d5 = d4 * d, d6 = d5 * d, d7 = d6 * d;
  float env = 1.0f / d + ENV_A * d5 + ENV_B * d6 + ENV_C * d7;
  #pragma unroll
  for (int n = 0; n < NRAD; ++n)
    rbf[e * NRAD + n] = env * sinf(freq[n] * d);
  for (int l = 0; l < LMAX; ++l)
    for (int n = 0; n < NRAD; ++n) {
      float x = d * zj[l * NRAD + n];
      rbfenv[e * (LMAX * NRAD) + l * NRAD + n] =
          env * bnorm[l * NRAD + n] * sph_jn(l, x);
    }
}

// ----------------------- triplet angular bases -----------------------------
__global__ __launch_bounds__(256) void dn_cbf(
    const float* __restrict__ R, const int* __restrict__ i3,
    const int* __restrict__ j3, const int* __restrict__ k3,
    float* __restrict__ cbf) {
  long t = (long)blockIdx.x * 256 + threadIdx.x;
  if (t >= T_TRIP) return;
  int ai = i3[t], aj = j3[t], ak = k3[t];
  float r1x = R[aj * 3 + 0] - R[ai * 3 + 0];
  float r1y = R[aj * 3 + 1] - R[ai * 3 + 1];
  float r1z = R[aj * 3 + 2] - R[ai * 3 + 2];
  float r2x = R[ak * 3 + 0] - R[aj * 3 + 0];
  float r2y = R[ak * 3 + 1] - R[aj * 3 + 1];
  float r2z = R[ak * 3 + 2] - R[aj * 3 + 2];
  float cx = r1y * r2z - r1z * r2y;
  float cy = r1z * r2x - r1x * r2z;
  float cz = r1x * r2y - r1y * r2x;
  float cn = sqrtf(cx * cx + cy * cy + cz * cz);
  float dp = r1x * r2x + r1y * r2y + r1z * r2z;
  float ca = cosf(atan2f(cn, dp));
  const float inv4pi = 0.07957747154594767f;
  float Pm = 1.0f, Pc = ca;
  cbf[t * LMAX + 0] = sqrtf(1.0f * inv4pi);
  cbf[t * LMAX + 1] = sqrtf(3.0f * inv4pi) * Pc;
  #pragma unroll
  for (int l = 1; l <= LMAX - 2; ++l) {
    float Pn = ((2.0f * l + 1.0f) * ca * Pc - (float)l * Pm) / (float)(l + 1);
    Pm = Pc; Pc = Pn;
    cbf[t * LMAX + l + 1] = sqrtf((2.0f * (l + 1) + 1.0f) * inv4pi) * Pc;
  }
}

// ----------------------- atom-embedding gather -----------------------------
__global__ __launch_bounds__(256) void dn_gather_embed(
    const float* __restrict__ atom, const int* __restrict__ Z,
    const int* __restrict__ idx, float* __restrict__ out) {
  long g = (long)blockIdx.x * 256 + threadIdx.x;  // e*128 + c
  if (g >= (long)E_EDGES * EMB) return;
  long e = g >> 7;
  int  c = (int)(g & 127);
  out[g] = atom[(long)Z[idx[e]] * EMB + c];
}

// ----------------------- rbf -> EMB dense (K=6) ----------------------------
__global__ __launch_bounds__(256) void dn_rbf_dense(
    const float* __restrict__ rbf, const float* __restrict__ W,
    const float* __restrict__ b, float* __restrict__ out) {
  long g = (long)blockIdx.x * 256 + threadIdx.x;
  if (g >= (long)E_EDGES * EMB) return;
  long e = g >> 7;
  int  c = (int)(g & 127);
  float s = b[c];
  #pragma unroll
  for (int n = 0; n < NRAD; ++n) s += rbf[e * NRAD + n] * W[n * EMB + c];
  out[g] = swishf(s);
}

// ------------- xb *= (rbf @ rbf1) @ rbf2   (K=6->8->128) -------------------
__global__ __launch_bounds__(256) void dn_edge_mul_g(
    const float* __restrict__ rbf, const float* __restrict__ rbf1,
    const float* __restrict__ rbf2, float* __restrict__ xb) {
  long g = (long)blockIdx.x * 256 + threadIdx.x;
  if (g >= (long)E_EDGES * EMB) return;
  long e = g >> 7;
  int  c = (int)(g & 127);
  float t8[BASIS];
  #pragma unroll
  for (int h = 0; h < BASIS; ++h) {
    float s = 0.0f;
    #pragma unroll
    for (int n = 0; n < NRAD; ++n) s += rbf[e * NRAD + n] * rbf1[n * BASIS + h];
    t8[h] = s;
  }
  float gg = 0.0f;
  #pragma unroll
  for (int h = 0; h < BASIS; ++h) gg += t8[h] * rbf2[h * EMB + c];
  xb[g] *= gg;
}

// -------- triplet fusion: sbf -> 8 -> 64, gather xd, scatter-add -----------
__global__ __launch_bounds__(256) void dn_triplet(
    const float* __restrict__ cbf, const float* __restrict__ rbfenv,
    const float* __restrict__ xd, const int* __restrict__ ekj_idx,
    const int* __restrict__ eji_idx, const float* __restrict__ sbf1,
    const float* __restrict__ sbf2, float* __restrict__ acc) {
  long t = (long)blockIdx.x * 256 + threadIdx.x;
  if (t >= T_TRIP) return;
  int ekj = ekj_idx[t];
  int eji = eji_idx[t];
  float cb[LMAX];
  #pragma unroll
  for (int l = 0; l < LMAX; ++l) cb[l] = cbf[t * LMAX + l];
  float sb[LMAX * NRAD];
  #pragma unroll
  for (int l = 0; l < LMAX; ++l)
    #pragma unroll
    for (int n = 0; n < NRAD; ++n)
      sb[l * NRAD + n] = rbfenv[(long)ekj * (LMAX * NRAD) + l * NRAD + n] * cb[l];
  float s8[BASIS];
  #pragma unroll
  for (int h = 0; h < BASIS; ++h) {
    float s = 0.0f;
    #pragma unroll
    for (int q = 0; q < LMAX * NRAD; ++q) s += sb[q] * sbf1[q * BASIS + h];
    s8[h] = s;
  }
  #pragma unroll
  for (int c = 0; c < INTEMB; ++c) {
    float v = 0.0f;
    #pragma unroll
    for (int h = 0; h < BASIS; ++h) v += s8[h] * sbf2[h * INTEMB + c];
    v *= xd[(long)ekj * INTEMB + c];
    atomicAdd(&acc[(long)eji * INTEMB + c], v);
  }
}

// ------- out-block front: atom_t += (rbf @ o.rbf) * x  scattered by i ------
__global__ __launch_bounds__(256) void dn_out_gather(
    const float* __restrict__ rbf, const float* __restrict__ orbf,
    const int* __restrict__ idnb_i, const float* __restrict__ x,
    float* __restrict__ atom_t) {
  long g = (long)blockIdx.x * 256 + threadIdx.x;
  if (g >= (long)E_EDGES * EMB) return;
  long e = g >> 7;
  int  c = (int)(g & 127);
  float s = 0.0f;
  #pragma unroll
  for (int n = 0; n < NRAD; ++n) s += rbf[e * NRAD + n] * orbf[n * EMB + c];
  atomicAdd(&atom_t[(long)idnb_i[e] * EMB + c], s * x[g]);
}

// ----------------- per-molecule reduction of P -----------------------------
__global__ __launch_bounds__(256) void dn_mol_reduce(
    const float* __restrict__ P, const int* __restrict__ batch_seg,
    float* __restrict__ out) {
  long g = (long)blockIdx.x * 256 + threadIdx.x;  // a*12 + c
  if (g >= (long)A_ATOMS * NLBL) return;
  long a = g / NLBL;
  int  c = (int)(g % NLBL);
  atomicAdd(&out[(long)batch_seg[a] * NLBL + c], P[g]);
}

// ---------------------------------------------------------------------------
// WMMA GEMM: C = f(A[M,K] @ W[K,N] (+bias)), f per (mode, act):
//   mode 0: C = act(AW + b)
//   mode 1: C = C_old + act(AW + b)
//   mode 2: C = act(C_old + AW + b)
// Tile 128 x BNT (BNT = 64 or 128), BK = 32, 256 threads = 8 waves.
// Each wave computes 16 rows x BNT cols (BNT/16 WMMA tiles per k-step).
// A staged via async global->LDS (f32) when available, else sync f32->f16.
// Requires M % 128 == 0, K % 32 == 0 (true for all call sites here).
// ---------------------------------------------------------------------------
#define GBM 128
#define GBK 32
#define LDSB_S 40  // f16 elements per Bs row (32 + 8 pad)
#define LDSA_S 36  // f32 elements per As32 row (32 + 4 pad)

template <int BNT>
__global__ __launch_bounds__(256) void dn_gemm(
    const float* __restrict__ A, const float* __restrict__ W,
    const float* __restrict__ bias, float* __restrict__ C,
    int M, int N, int K, int mode, int act) {
  constexpr int NSUB  = BNT / 16;   // WMMA tiles per wave per k-step
  constexpr int CHUNK = BNT / 8;    // B f32 elements staged per thread
#if DN_ASYNC_OK
  __shared__ float    As32[GBM * LDSA_S];
#else
  __shared__ _Float16 As16[GBM * LDSB_S];
#endif
  __shared__ _Float16 Bs[BNT * LDSB_S];
  const int tid  = threadIdx.x;
  const int wave = tid >> 5;
  const int lane = tid & 31;
  const int half = lane >> 4;
  const int lm   = lane & 15;
  const long bm  = (long)blockIdx.x * GBM;
  const int  bn  = blockIdx.y * BNT;

  v8f acc[NSUB];
  #pragma unroll
  for (int nt = 0; nt < NSUB; ++nt)
    #pragma unroll
    for (int r = 0; r < 8; ++r) acc[nt][r] = 0.0f;

  for (int k0 = 0; k0 < K; k0 += GBK) {
    // ---- stage A tile (128 x 32 f32) ----
#if DN_ASYNC_OK
    #pragma unroll
    for (int i = 0; i < 4; ++i) {
      int idx = tid + i * 256;
      int r   = idx >> 3;
      int c4  = (idx & 7) << 2;
      __builtin_amdgcn_global_load_async_to_lds_b128(
          (dn_gv4i*)(A + (bm + r) * (long)K + k0 + c4),
          (dn_lv4i*)&As32[r * LDSA_S + c4], 0, 0);
    }
#else
    float4 va[4];
    #pragma unroll
    for (int i = 0; i < 4; ++i) {
      int idx = tid + i * 256;
      va[i] = *(const float4*)(A + (bm + (idx >> 3)) * (long)K + k0 + ((idx & 7) << 2));
    }
    #pragma unroll
    for (int i = 0; i < 4; ++i) {
      int idx = tid + i * 256;
      _Float16* dst = &As16[(idx >> 3) * LDSB_S + ((idx & 7) << 2)];
      dst[0] = (_Float16)va[i].x; dst[1] = (_Float16)va[i].y;
      dst[2] = (_Float16)va[i].z; dst[3] = (_Float16)va[i].w;
    }
#endif
    // ---- stage W tile transposed: Bs[n][k] = W[k0+k][bn+n], branchless pad ----
    {
      int   n    = tid % BNT;
      int   kk   = (tid / BNT) * CHUNK;
      float mask = (bn + n < N) ? 1.0f : 0.0f;
      int   nc   = (bn + n < N) ? (bn + n) : (N - 1);
      float wb[CHUNK];
      #pragma unroll
      for (int i = 0; i < CHUNK; ++i)
        wb[i] = W[(long)(k0 + kk + i) * N + nc] * mask;
      #pragma unroll
      for (int i = 0; i < CHUNK; ++i)
        Bs[n * LDSB_S + kk + i] = (_Float16)wb[i];
    }
    // prefetch next A k-tile into GL2 (gfx1250 global_prefetch_b8)
    if (k0 + GBK < K)
      __builtin_prefetch(A + (bm + (tid >> 1)) * (long)K + k0 + GBK + (tid & 1) * 16, 0, 0);
#if DN_ASYNC_OK
    __builtin_amdgcn_s_wait_asynccnt(0);
#endif
    __syncthreads();

    // ---- A fragment: lane holds row (wave*16+lm); K split by lane-half ----
    union { v16h v; _Float16 h[16]; unsigned u[8]; } Af;
#if DN_ASYNC_OK
    const float* Ar = &As32[(wave * 16 + lm) * LDSA_S];
    #pragma unroll
    for (int p = 0; p < 8; ++p) {
      int kb = ((p < 4) ? (p << 1) : (16 + ((p - 4) << 1))) + half * 8;
      Af.h[2 * p]     = (_Float16)Ar[kb];
      Af.h[2 * p + 1] = (_Float16)Ar[kb + 1];
    }
#else
    const unsigned* Arow = (const unsigned*)&As16[(wave * 16 + lm) * LDSB_S];
    #pragma unroll
    for (int p = 0; p < 8; ++p) {
      int kb = ((p < 4) ? (p << 1) : (16 + ((p - 4) << 1))) + half * 8;
      Af.u[p] = Arow[kb >> 1];
    }
#endif
    #pragma unroll
    for (int nt = 0; nt < NSUB; ++nt) {
      // B fragment: lane holds col (nt*16+lm); K = half*8 + e (e<8), 16 + half*8 + (e-8) (e>=8)
      union { v16h v; unsigned u[8]; } Bf;
      const unsigned* Brow = (const unsigned*)&Bs[(nt * 16 + lm) * LDSB_S];
      #pragma unroll
      for (int p = 0; p < 8; ++p)
        Bf.u[p] = Brow[(((p < 4) ? (p << 1) : (16 + ((p - 4) << 1))) + half * 8) >> 1];
      acc[nt] = __builtin_amdgcn_wmma_f32_16x16x32_f16(
          false, Af.v, false, Bf.v, (short)0, acc[nt], false, false);
    }
    __syncthreads();
  }

  // ---- epilogue: C/D layout -> row = r + half*8, col = lm per 16x16 tile ----
  #pragma unroll
  for (int nt = 0; nt < NSUB; ++nt) {
    int col = bn + nt * 16 + lm;
    if (col >= N) continue;
    float bv = bias ? bias[col] : 0.0f;
    #pragma unroll
    for (int r = 0; r < 8; ++r) {
      long row = bm + wave * 16 + half * 8 + r;
      long off = row * (long)N + col;
      float s = acc[nt][r] + bv;
      if (mode == 2) s += C[off];
      if (act) s = swishf(s);
      if (mode == 1) s += C[off];
      C[off] = s;
    }
  }
}

// ---------------------------------------------------------------------------
extern "C" void kernel_launch(void* const* d_in, const int* in_sizes, int n_in,
                              void* d_out, int out_size, void* d_ws, size_t ws_size,
                              hipStream_t stream) {
  auto F = [&](int i) { return (const float*)d_in[i]; };
  auto I = [&](int i) { return (const int*)d_in[i]; };

  // graph inputs
  const int*   Z        = I(0);
  const float* R        = F(1);
  const int*   batch    = I(2);
  const int*   idnb_i   = I(3);
  const int*   idnb_j   = I(4);
  const int*   id_exp   = I(5);
  const int*   id_red   = I(6);
  const int*   id3i     = I(7);
  const int*   id3j     = I(8);
  const int*   id3k     = I(9);
  // params, JAX pytree order (sorted dict keys, lists in order)
  const float* emb_atom = F(10);
  const float* emb_b    = F(11);
  const float* emb_rbfb = F(12);
  const float* emb_rbfw = F(13);
  const float* emb_w    = F(14);  // [384,128]
  const float* freq     = F(15);
  // int block b: base = 16 + b*24 (leaf order: after0 b1,b2,w1,w2; after1 ...;
  //   before0 ...; down, fin_b, fin_w, ji_b, ji_w, kj_b, kj_w, rbf1, rbf2,
  //   sbf1, sbf2, up)
  // out block o: base = 112 + o*9 (dense0 b,w; dense1 b,w; dense2 b,w;
  //   final, rbf, up)
  (void)n_in; (void)in_sizes; (void)ws_size;

  // workspace layout
  size_t off = 0;
  auto WS = [&](size_t n) { float* p = (float*)d_ws + off; off += n; return p; };
  float* zj     = WS(64);
  float* bnorm  = WS(64);
  float* rbf    = WS((size_t)E_EDGES * NRAD);
  float* rbfenv = WS((size_t)E_EDGES * LMAX * NRAD);
  float* cbf    = WS((size_t)T_TRIP * LMAX);
  float* x      = WS((size_t)E_EDGES * EMB);
  float* xa     = WS((size_t)E_EDGES * EMB);
  float* xb     = WS((size_t)E_EDGES * EMB);
  float* h1     = WS((size_t)E_EDGES * EMB);
  float* xd     = WS((size_t)E_EDGES * INTEMB);
  float* acc    = WS((size_t)E_EDGES * INTEMB);
  float* atom_t = WS((size_t)A_ATOMS * EMB);
  float* atom_u = WS((size_t)A_ATOMS * OUTEMB);
  float* atom_v = WS((size_t)A_ATOMS * OUTEMB);
  float* P      = WS((size_t)A_ATOMS * NLBL);

  auto zero = [&](float* p, long n) {
    dn_zero<<<(unsigned)((n + 255) / 256), 256, 0, stream>>>(p, n);
  };
  auto gemm = [&](const float* Ap, const float* Wt, const float* b, float* Cp,
                  int M, int N, int K, int mode, int act) {
    if (N % 128 == 0) {
      dim3 grid((unsigned)(M / GBM), (unsigned)(N / 128));
      dn_gemm<128><<<grid, 256, 0, stream>>>(Ap, Wt, b, Cp, M, N, K, mode, act);
    } else {
      dim3 grid((unsigned)(M / GBM), (unsigned)((N + 63) / 64));
      dn_gemm<64><<<grid, 256, 0, stream>>>(Ap, Wt, b, Cp, M, N, K, mode, act);
    }
  };

  const unsigned gEC = (unsigned)(((long)E_EDGES * EMB) / 256);  // per-(e,c) grids
  const unsigned gT  = (unsigned)(T_TRIP / 256);

  // ---- geometric bases ----
  dn_bessel_init<<<1, 32, 0, stream>>>(zj, bnorm);
  dn_edge_basis<<<E_EDGES / 256, 256, 0, stream>>>(R, idnb_i, idnb_j, freq,
                                                   zj, bnorm, rbf, rbfenv);
  dn_cbf<<<gT, 256, 0, stream>>>(R, id3i, id3j, id3k, cbf);

  // ---- embedding block: x = swish([a_i | a_j | rbf_e] @ W + b) ----
  dn_gather_embed<<<gEC, 256, 0, stream>>>(emb_atom, Z, idnb_i, xa);
  dn_gather_embed<<<gEC, 256, 0, stream>>>(emb_atom, Z, idnb_j, xb);
  dn_rbf_dense<<<gEC, 256, 0, stream>>>(rbf, emb_rbfw, emb_rbfb, h1);
  gemm(xa, emb_w,             nullptr, x, E_EDGES, EMB, EMB, 0, 0);
  gemm(xb, emb_w + 128 * EMB, nullptr, x, E_EDGES, EMB, EMB, 1, 0);
  gemm(h1, emb_w + 256 * EMB, emb_b,   x, E_EDGES, EMB, EMB, 2, 1);

  zero(P, (long)A_ATOMS * NLBL);

  auto out_block = [&](int o) {
    int ob = 112 + o * 9;
    zero(atom_t, (long)A_ATOMS * EMB);
    dn_out_gather<<<gEC, 256, 0, stream>>>(rbf, F(ob + 7), idnb_i, x, atom_t);
    gemm(atom_t, F(ob + 8), nullptr,   atom_u, A_ATOMS, OUTEMB, EMB,    0, 0);
    gemm(atom_u, F(ob + 1), F(ob + 0), atom_v, A_ATOMS, OUTEMB, OUTEMB, 0, 1);
    gemm(atom_v, F(ob + 3), F(ob + 2), atom_u, A_ATOMS, OUTEMB, OUTEMB, 0, 1);
    gemm(atom_u, F(ob + 5), F(ob + 4), atom_v, A_ATOMS, OUTEMB, OUTEMB, 0, 1);
    gemm(atom_v, F(ob + 6), nullptr,   P,      A_ATOMS, NLBL,   OUTEMB, 1, 0);
  };

  out_block(0);

  for (int b = 0; b < 4; ++b) {
    int ib = 16 + b * 24;
    // x_ji / x_kj
    gemm(x, F(ib + 16), F(ib + 15), xa, E_EDGES, EMB, EMB, 0, 1);  // ji
    gemm(x, F(ib + 18), F(ib + 17), xb, E_EDGES, EMB, EMB, 0, 1);  // kj
    dn_edge_mul_g<<<gEC, 256, 0, stream>>>(rbf, F(ib + 19), F(ib + 20), xb);
    gemm(xb, F(ib + 12), nullptr, xd, E_EDGES, INTEMB, EMB, 0, 1); // down
    // triplet message passing: acc = segsum(x_kj[ekj] * sbf2t, eji)
    zero(acc, (long)E_EDGES * INTEMB);
    dn_triplet<<<gT, 256, 0, stream>>>(cbf, rbfenv, xd, id_exp, id_red,
                                       F(ib + 21), F(ib + 22), acc);
    gemm(acc, F(ib + 23), nullptr, xa, E_EDGES, EMB, INTEMB, 1, 1); // xa += swish(acc@up)
    // before-residual (1)
    gemm(xa, F(ib + 10), F(ib + 8), h1, E_EDGES, EMB, EMB, 0, 1);
    gemm(h1, F(ib + 11), F(ib + 9), xa, E_EDGES, EMB, EMB, 1, 1);
    // x = x + swish(xa @ fin_w + fin_b)
    gemm(xa, F(ib + 14), F(ib + 13), x, E_EDGES, EMB, EMB, 1, 1);
    // after-residuals (2)
    gemm(x,  F(ib + 2), F(ib + 0), h1, E_EDGES, EMB, EMB, 0, 1);
    gemm(h1, F(ib + 3), F(ib + 1), x,  E_EDGES, EMB, EMB, 1, 1);
    gemm(x,  F(ib + 6), F(ib + 4), h1, E_EDGES, EMB, EMB, 0, 1);
    gemm(h1, F(ib + 7), F(ib + 5), x,  E_EDGES, EMB, EMB, 1, 1);
    out_block(b + 1);
  }

  // ---- per-molecule reduction ----
  zero((float*)d_out, out_size);
  dn_mol_reduce<<<(unsigned)(((long)A_ATOMS * NLBL + 255) / 256), 256, 0, stream>>>(
      P, batch, (float*)d_out);
}
